// FluxJointAttention_63273458205327
// MI455X (gfx1250) — matrix operand order repaired and checked by reading, not compiled, not run on hardware
//
#include <hip/hip_runtime.h>
#include <hip/hip_bf16.h>

typedef unsigned short u16;
typedef unsigned int   u32;
typedef __attribute__((ext_vector_type(8)))  __bf16 v8bf;
typedef __attribute__((ext_vector_type(16))) __bf16 v16bf;
typedef __attribute__((ext_vector_type(8)))  float  v8f;

constexpr int kH   = 24;
constexpr int kDH  = 128;
constexpr int kD   = 3072;
constexpr int kSA  = 2048;
constexpr int kSB  = 512;
constexpr int kS   = 2560;     // kSA + kSB
constexpr int kQKV = 9216;     // 3*kD

// ---------- helpers ----------

__device__ __forceinline__ u16 f2bf(float f) {
  u32 u = __builtin_bit_cast(u32, f);
  u32 r = (u + 0x7FFFu + ((u >> 16) & 1u)) >> 16;   // round-to-nearest-even
  return (u16)r;
}

// Build a 16-element bf16 fragment from two 16-byte chunks (ds/global b128 loads).
__device__ __forceinline__ v16bf frag_from(const u16* p0, const u16* p1) {
  v8bf lo = *reinterpret_cast<const v8bf*>(p0);
  v8bf hi = *reinterpret_cast<const v8bf*>(p1);
  return __builtin_shufflevector(lo, hi, 0,1,2,3,4,5,6,7,8,9,10,11,12,13,14,15);
}

__device__ __forceinline__ v8f wmma_bf16(v16bf a, v16bf b, v8f c) {
  return __builtin_amdgcn_wmma_f32_16x16x32_bf16(
      /*neg_a=*/false, a, /*neg_b=*/false, b,
      /*c_mod=*/(short)0, c, /*reuse_a=*/false, /*reuse_b=*/false);
}

__device__ __forceinline__ v8f vzero8() {
  v8f z;
  #pragma unroll
  for (int i = 0; i < 8; ++i) z[i] = 0.0f;
  return z;
}

// CDNA5 async DMA: global (16B/lane) -> LDS, tracked by ASYNCcnt.
__device__ __forceinline__ void async_copy_b128(u32 lds_off, const u16* gaddr) {
  asm volatile("global_load_async_to_lds_b128 %0, %1, off"
               :: "v"(lds_off), "v"(gaddr) : "memory");
}
__device__ __forceinline__ void wait_asynccnt0() {
  asm volatile("s_wait_asynccnt 0x0" ::: "memory");
}

// ---------- Kernel 0: fp32 -> bf16 bulk cast (weights / activations), float4 vectorized ----------

__global__ __launch_bounds__(256) void cvt_bf16_kernel(
    const float* __restrict__ in, u16* __restrict__ out, int n4)
{
  const int i = blockIdx.x * 256 + threadIdx.x;
  if (i < n4) {
    float4 f = reinterpret_cast<const float4*>(in)[i];
    u32 lo = (u32)f2bf(f.x) | ((u32)f2bf(f.y) << 16);
    u32 hi = (u32)f2bf(f.z) | ((u32)f2bf(f.w) << 16);
    reinterpret_cast<uint2*>(out)[i] = make_uint2(lo, hi);
  }
}

// ---------- Kernel 1/4: C[M,N] = A[M,K] * B[N,K]^T + bias[N] ----------
// bf16 in, fp32 out. Block tile 128x256, 256 threads = 8 waves (2x4), wave tile 64x64
// = 4x4 WMMA tiles. k-step 32. Double-buffered async global->LDS DMA pipeline.

constexpr int LSTR = 56;  // LDS row stride in ushorts (conflict-free, 16B aligned)

__global__ __launch_bounds__(256) void gemm_bf16_kernel(
    const u16* __restrict__ A, const u16* __restrict__ Bw,
    const float* __restrict__ bias, float* __restrict__ C,
    int M, int N, int K)
{
  __shared__ u16 As[2][128 * LSTR];
  __shared__ u16 Bs[2][256 * LSTR];

  const int tid  = threadIdx.x;
  const int lane = tid & 31;
  const int w    = tid >> 5;
  const int wm   = w & 1;           // 0..1 -> M half
  const int wn   = w >> 1;          // 0..3 -> N quarter
  const int r    = lane & 15;
  const int kh   = lane >> 4;

  const int m0 = blockIdx.y * 128;
  const int n0 = blockIdx.x * 256;

  // staging assignment: A tile 128x32 -> 2 threads/row; B tile 256x32 -> 1 thread/row
  const int ra = tid >> 1;
  const int ca = (tid & 1) * 16;

  u32 lA[2], lB[2];
  #pragma unroll
  for (int b = 0; b < 2; ++b) {
    lA[b] = (u32)(size_t)&As[b][ra * LSTR + ca];
    lB[b] = (u32)(size_t)&Bs[b][tid * LSTR];
  }
  const u16* apBase = A  + (size_t)(m0 + ra) * K + ca;
  const u16* bpBase = Bw + (size_t)(n0 + tid) * K;

  v8f acc[4][4];
  #pragma unroll
  for (int i = 0; i < 4; ++i)
    #pragma unroll
    for (int j = 0; j < 4; ++j) acc[i][j] = vzero8();

  // prologue: stage k-tile 0 into buffer 0
  {
    const u16* ap = apBase;
    const u16* bp = bpBase;
    async_copy_b128(lA[0],      ap);
    async_copy_b128(lA[0] + 16, ap + 8);
    async_copy_b128(lB[0],      bp);
    async_copy_b128(lB[0] + 16, bp + 8);
    async_copy_b128(lB[0] + 32, bp + 16);
    async_copy_b128(lB[0] + 48, bp + 24);
  }

  int buf = 0;
  for (int k0 = 0; k0 < K; k0 += 32) {
    wait_asynccnt0();     // copies for this k-tile complete (issued one iter ago)
    __syncthreads();      // ...in every wave; also: everyone done reading buf^1

    if (k0 + 32 < K) {    // overlap: DMA next tile into the other buffer
      const int nb = buf ^ 1;
      const u16* ap = apBase + k0 + 32;
      const u16* bp = bpBase + k0 + 32;
      async_copy_b128(lA[nb],      ap);
      async_copy_b128(lA[nb] + 16, ap + 8);
      async_copy_b128(lB[nb],      bp);
      async_copy_b128(lB[nb] + 16, bp + 8);
      async_copy_b128(lB[nb] + 32, bp + 16);
      async_copy_b128(lB[nb] + 48, bp + 24);
      __builtin_prefetch(ap + 32, 0, 1);   // pull k-tile after next into L2
      __builtin_prefetch(bp + 32, 0, 1);
    }

    // ---- compute current tile ----
    v16bf af[4];
    #pragma unroll
    for (int mt = 0; mt < 4; ++mt) {
      const u16* p = &As[buf][(wm * 64 + mt * 16 + r) * LSTR + kh * 8];
      af[mt] = frag_from(p, p + 16);       // A layout: k halves of 8
    }
    #pragma unroll
    for (int nt = 0; nt < 4; ++nt) {
      const u16* p = &Bs[buf][(wn * 64 + nt * 16 + r) * LSTR + kh * 8];
      v16bf bfr = frag_from(p, p + 16);    // B layout: k halves of 8
      #pragma unroll
      for (int mt = 0; mt < 4; ++mt)
        acc[mt][nt] = wmma_bf16(af[mt], bfr, acc[mt][nt]);
    }
    buf ^= 1;
  }

  // ---- epilogue: bias + store (C fragment: row = v + 8*kh, col = lane&15) ----
  #pragma unroll
  for (int nt = 0; nt < 4; ++nt) {
    const int col = n0 + wn * 64 + nt * 16 + r;
    const float bv = bias[col];
    #pragma unroll
    for (int mt = 0; mt < 4; ++mt) {
      const int rb = m0 + wm * 64 + mt * 16 + 8 * kh;
      #pragma unroll
      for (int v = 0; v < 8; ++v)
        C[(size_t)(rb + v) * N + col] = acc[mt][nt][v] + bv;
    }
  }
}

// ---------- Kernel 2: per-(s,h) RMS-norm(q,k) + RoPE + reorder to bf16 q/k [h][s][dh], v^T [h][dh][s] ----------

__global__ __launch_bounds__(128) void norm_rope_kernel(
    const float* __restrict__ qkv_a, const float* __restrict__ qkv_b,
    const float* __restrict__ rope,
    const float* __restrict__ wqa, const float* __restrict__ wka,
    const float* __restrict__ wqb, const float* __restrict__ wkb,
    u16* __restrict__ qg, u16* __restrict__ kg, u16* __restrict__ vtg)
{
  const int s = blockIdx.x;        // 0..kS-1 (b rows first, then a rows)
  const int h = blockIdx.y;
  const int t = threadIdx.x;       // 0..127

  const float* row; const float* wq; const float* wk;
  if (s < kSB) { row = qkv_b + (size_t)s * kQKV;          wq = wqb; wk = wkb; }
  else         { row = qkv_a + (size_t)(s - kSB) * kQKV;  wq = wqa; wk = wka; }

  const float qv = row[h * kDH + t];
  const float kv = row[kD  + h * kDH + t];
  const float vv = row[2*kD + h * kDH + t];

  __shared__ float rq[128], rk[128];
  rq[t] = qv * qv; rk[t] = kv * kv;
  __syncthreads();
  #pragma unroll
  for (int off = 64; off > 0; off >>= 1) {
    if (t < off) { rq[t] += rq[t + off]; rk[t] += rk[t + off]; }
    __syncthreads();
  }
  const float qn = qv * rsqrtf(rq[0] * (1.0f / kDH) + 1e-6f) * wq[t];
  const float kn = kv * rsqrtf(rk[0] * (1.0f / kDH) + 1e-6f) * wk[t];

  // RoPE: out[2p] = c*x[2p] - s*x[2p+1]; out[2p+1] = s*x[2p] + c*x[2p+1]
  const int p = t >> 1, i = t & 1;
  const float f0 = rope[(size_t)s * 256 + p * 4 + i * 2 + 0];
  const float f1 = rope[(size_t)s * 256 + p * 4 + i * 2 + 1];
  const float qo = __shfl_xor(qn, 1, 32);   // pair partner in same wave
  const float ko = __shfl_xor(kn, 1, 32);
  const float qe = i ? qo : qn, qod = i ? qn : qo;
  const float ke = i ? ko : kn, kod = i ? kn : ko;
  const float qr = f0 * qe + f1 * qod;
  const float kr = f0 * ke + f1 * kod;

  const size_t idx = ((size_t)h * kS + s) * kDH + t;
  qg[idx] = f2bf(qr);
  kg[idx] = f2bf(kr);
  vtg[((size_t)h * kDH + t) * kS + s] = f2bf(vv);
}

// ---------- Kernel 3: flash attention, one head x 128 query rows per block, 8 waves x 16 rows ----------

constexpr int PSTR = 72;   // per-wave P-tile LDS row stride (ushorts), conflict-free & 16B aligned

__global__ __launch_bounds__(256) void attn_kernel(
    const u16* __restrict__ qg, const u16* __restrict__ kg,
    const u16* __restrict__ vtg, u16* __restrict__ o)
{
  __shared__ u16 Pl[8 * 16 * PSTR];   // per-wave private 16x64 bf16 P tiles

  const int tid  = threadIdx.x;
  const int lane = tid & 31;
  const int w    = tid >> 5;
  const int r    = lane & 15;
  const int kh   = lane >> 4;
  const int h    = blockIdx.y;
  const int q0   = blockIdx.x * 128 + w * 16;

  const float scale = 0.08838834764831845f;   // 1/sqrt(128)

  // Q fragments resident: 4 k-steps over DH=128 (direct global b128 loads)
  v16bf aq[4];
  {
    const u16* qrow = qg + ((size_t)h * kS + q0 + r) * kDH;
    #pragma unroll
    for (int ks = 0; ks < 4; ++ks) {
      const u16* p = qrow + ks * 32 + kh * 8;
      aq[ks] = frag_from(p, p + 16);
    }
  }

  float mrow[8], lrow[8];
  v8f oacc[8];
  #pragma unroll
  for (int v = 0; v < 8; ++v) { mrow[v] = -1e30f; lrow[v] = 0.0f; oacc[v] = vzero8(); }

  u16* myP = &Pl[w * 16 * PSTR];

  for (int kb0 = 0; kb0 < kS; kb0 += 64) {
    // ---- scores: 16 rows x 64 keys ----
    v8f sacc[4];
    #pragma unroll
    for (int j = 0; j < 4; ++j) sacc[j] = vzero8();
    #pragma unroll
    for (int ks = 0; ks < 4; ++ks) {
      #pragma unroll
      for (int j = 0; j < 4; ++j) {
        const u16* p = kg + ((size_t)h * kS + kb0 + j * 16 + r) * kDH + ks * 32 + kh * 8;
        v16bf bk = frag_from(p, p + 16);
        sacc[j] = wmma_bf16(aq[ks], bk, sacc[j]);
      }
    }

    // ---- online softmax (row = v + 8*kh; values spread over 16 lanes x 4 tiles) ----
    float mnew[8], corr[8];
    #pragma unroll
    for (int v = 0; v < 8; ++v) {
      float m = fmaxf(fmaxf(sacc[0][v], sacc[1][v]), fmaxf(sacc[2][v], sacc[3][v]));
      #pragma unroll
      for (int msk = 1; msk < 16; msk <<= 1) m = fmaxf(m, __shfl_xor(m, msk, 32));
      m *= scale;
      mnew[v] = fmaxf(mrow[v], m);
      corr[v] = __expf(mrow[v] - mnew[v]);
    }

    float rs[8];
    #pragma unroll
    for (int v = 0; v < 8; ++v) rs[v] = 0.0f;
    #pragma unroll
    for (int j = 0; j < 4; ++j)
      #pragma unroll
      for (int v = 0; v < 8; ++v) {
        const float pv = __expf(sacc[j][v] * scale - mnew[v]);
        rs[v] += pv;
        myP[(v + 8 * kh) * PSTR + j * 16 + r] = f2bf(pv);   // transpose via LDS
      }
    #pragma unroll
    for (int v = 0; v < 8; ++v) {
      float t2 = rs[v];
      #pragma unroll
      for (int msk = 1; msk < 16; msk <<= 1) t2 += __shfl_xor(t2, msk, 32);
      lrow[v] = lrow[v] * corr[v] + t2;
      mrow[v] = mnew[v];
    }
    #pragma unroll
    for (int jj = 0; jj < 8; ++jj)
      #pragma unroll
      for (int v = 0; v < 8; ++v) oacc[jj][v] *= corr[v];

    // ---- O += P @ V (P from private LDS as A-frags; V^T direct from global) ----
    #pragma unroll
    for (int ks = 0; ks < 2; ++ks) {
      const u16* pp = &myP[r * PSTR + ks * 32 + kh * 8];
      v16bf ap = frag_from(pp, pp + 16);
      #pragma unroll
      for (int jj = 0; jj < 8; ++jj) {
        const u16* vp = vtg + ((size_t)h * kDH + jj * 16 + r) * kS + kb0 + ks * 32 + kh * 8;
        v16bf bv = frag_from(vp, vp + 16);
        oacc[jj] = wmma_bf16(ap, bv, oacc[jj]);
      }
    }
  }

  // ---- normalize + store O as bf16 (s, h*128+d) for the out-projection GEMM ----
  #pragma unroll
  for (int jj = 0; jj < 8; ++jj) {
    const int col = h * kDH + jj * 16 + r;
    #pragma unroll
    for (int v = 0; v < 8; ++v) {
      const int row = q0 + v + 8 * kh;
      o[(size_t)row * kD + col] = f2bf(oacc[jj][v] / lrow[v]);
    }
  }
}

// ---------- launch ----------

extern "C" void kernel_launch(void* const* d_in, const int* in_sizes, int n_in,
                              void* d_out, int out_size, void* d_ws, size_t ws_size,
                              hipStream_t stream)
{
  (void)in_sizes; (void)n_in; (void)out_size; (void)ws_size;

  const float* hs_a    = (const float*)d_in[0];
  const float* hs_b    = (const float*)d_in[1];
  const float* rope    = (const float*)d_in[2];
  const float* a_qkv_w = (const float*)d_in[3];
  const float* a_qkv_b = (const float*)d_in[4];
  const float* b_qkv_w = (const float*)d_in[5];
  const float* b_qkv_b = (const float*)d_in[6];
  const float* nqa     = (const float*)d_in[7];
  const float* nka     = (const float*)d_in[8];
  const float* nqb     = (const float*)d_in[9];
  const float* nkb     = (const float*)d_in[10];
  const float* a_out_w = (const float*)d_in[11];
  const float* a_out_b = (const float*)d_in[12];
  const float* b_out_w = (const float*)d_in[13];
  const float* b_out_b = (const float*)d_in[14];

  char* ws = (char*)d_ws;
  size_t off = 0;
  float* qkv_a  = (float*)(ws + off); off += (size_t)kSA * kQKV * 4;
  float* qkv_b  = (float*)(ws + off); off += (size_t)kSB * kQKV * 4;
  u16*   qbuf   = (u16*)  (ws + off); off += (size_t)kH * kS * kDH * 2;
  u16*   kbuf   = (u16*)  (ws + off); off += (size_t)kH * kS * kDH * 2;
  u16*   vtbuf  = (u16*)  (ws + off); off += (size_t)kH * kDH * kS * 2;
  u16*   obuf   = (u16*)  (ws + off); off += (size_t)kS * kD * 2;
  u16*   hsa_bf = (u16*)  (ws + off); off += (size_t)kSA * kD * 2;
  u16*   hsb_bf = (u16*)  (ws + off); off += (size_t)kSB * kD * 2;
  u16*   wqa_bf = (u16*)  (ws + off); off += (size_t)kQKV * kD * 2;
  u16*   wqb_bf = (u16*)  (ws + off); off += (size_t)kQKV * kD * 2;
  u16*   woa_bf = (u16*)  (ws + off); off += (size_t)kD * kD * 2;
  u16*   wob_bf = (u16*)  (ws + off); off += (size_t)kD * kD * 2;

  // 0) bulk fp32 -> bf16 casts (activations + all GEMM weights)
  auto cvt = [&](const float* src, u16* dst, size_t n) {
    const int n4 = (int)(n / 4);
    cvt_bf16_kernel<<<dim3((n4 + 255) / 256), 256, 0, stream>>>(src, dst, n4);
  };
  cvt(hs_a,    hsa_bf, (size_t)kSA * kD);
  cvt(hs_b,    hsb_bf, (size_t)kSB * kD);
  cvt(a_qkv_w, wqa_bf, (size_t)kQKV * kD);
  cvt(b_qkv_w, wqb_bf, (size_t)kQKV * kD);
  cvt(a_out_w, woa_bf, (size_t)kD * kD);
  cvt(b_out_w, wob_bf, (size_t)kD * kD);

  // 1) QKV projections (bf16 WMMA, double-buffered async-LDS pipeline)
  gemm_bf16_kernel<<<dim3(kQKV / 256, kSA / 128), 256, 0, stream>>>(
      hsa_bf, wqa_bf, a_qkv_b, qkv_a, kSA, kQKV, kD);
  gemm_bf16_kernel<<<dim3(kQKV / 256, kSB / 128), 256, 0, stream>>>(
      hsb_bf, wqb_bf, b_qkv_b, qkv_b, kSB, kQKV, kD);

  // 2) RMS-norm + RoPE + layout for attention
  norm_rope_kernel<<<dim3(kS, kH), 128, 0, stream>>>(
      qkv_a, qkv_b, rope, nqa, nka, nqb, nkb, qbuf, kbuf, vtbuf);

  // 3) Flash attention (writes bf16 O)
  attn_kernel<<<dim3(kS / 128, kH), 256, 0, stream>>>(qbuf, kbuf, vtbuf, obuf);

  // 4) Output projections (out_a first, then out_b, flat in d_out)
  float* out_a = (float*)d_out;
  float* out_b = out_a + (size_t)kSA * kD;
  gemm_bf16_kernel<<<dim3(kD / 256, kSA / 128), 256, 0, stream>>>(
      obuf + (size_t)kSB * kD, woa_bf, a_out_b, out_a, kSA, kD, kD);
  gemm_bf16_kernel<<<dim3(kD / 256, kSB / 128), 256, 0, stream>>>(
      obuf, wob_bf, b_out_b, out_b, kSB, kD, kD);
}
